// NeuroSAT_11330123727539
// MI455X (gfx1250) — matrix operand order repaired and checked by reading, not compiled, not run on hardware
//
#include <hip/hip_runtime.h>
#include <hip/hip_bf16.h>
#include <math.h>

typedef __bf16 bf16;
typedef __attribute__((ext_vector_type(16))) __bf16 bf16x16;
typedef __attribute__((ext_vector_type(8)))  __bf16 bf16x8;
typedef __attribute__((ext_vector_type(8)))  float   f32x8;

#define D_DIM     128
#define N_BATCH   32
#define NVARS     16000
#define NLITS     32000
#define NCLAUSES  64000
#define NEDGES    192000
#define NROUNDS   8

// ---------------------------------------------------------------------------
// WMMA helpers (CDNA5: v_wmma_f32_16x16x32_bf16, wave32)
// ---------------------------------------------------------------------------
static __device__ __forceinline__ f32x8 wmma_bf16(bf16x16 a, bf16x16 b, f32x8 c) {
  return __builtin_amdgcn_wmma_f32_16x16x32_bf16(false, a, false, b, (short)0, c,
                                                 false, false);
}

// A-matrix 16x32 bf16 fragment. Layout: lanes 0-15 hold row m, K = k0+{0..15};
// lanes 16-31 hold K = k0+{16..31}. -> two contiguous 16B loads per lane.
static __device__ __forceinline__ bf16x16 load_a_frag(const bf16* __restrict__ A,
                                                      int lda, int row0, int k0,
                                                      int lane) {
  int row = row0 + (lane & 15);
  int kb  = k0 + ((lane >> 4) << 4);
  const bf16* p = A + row * lda + kb;
  bf16x8 lo = *(const bf16x8*)(p);
  bf16x8 hi = *(const bf16x8*)(p + 8);
  bf16x16 r;
#pragma unroll
  for (int t = 0; t < 8; ++t) { r[t] = lo[t]; r[8 + t] = hi[t]; }
  return r;
}

// B-matrix 32x16 bf16 fragment from a [N][K] (pre-transposed) weight array.
// lanes 0-15: col n, K = k0..k0+15 ; lanes 16-31: col n, K = k0+16..k0+31.
// -> one contiguous 32B per-lane load.
static __device__ __forceinline__ bf16x16 load_b_frag(const bf16* __restrict__ Bt,
                                                      int ldb, int n0, int k0,
                                                      int lane) {
  int col = n0 + (lane & 15);
  int kb  = k0 + ((lane >> 4) << 4);
  return *(const bf16x16*)(Bt + col * ldb + kb);
}

static __device__ __forceinline__ float sigmoidf_(float x) {
  return 1.0f / (1.0f + __expf(-x));
}

// ---------------------------------------------------------------------------
// GEMM: Y[M,N] = act(A[M,K] @ Bt[N,K]^T + bias), bf16 in/out, f32 accumulate.
// Block = 128 threads = 4 waves; wave tile = 64 rows x 64 cols
// (16 wmma per K-step vs 16 b128 loads -> 1.0 load/wmma).
// grid = (M/256, N/64). M multiple of 256, N multiple of 64, K multiple of 32.
// ---------------------------------------------------------------------------
__global__ __launch_bounds__(128)
void gemm_bf16_kernel(const bf16* __restrict__ A, const bf16* __restrict__ Bt,
                      const float* __restrict__ bias, bf16* __restrict__ Y,
                      int M, int K, int N, int relu) {
  const int lane = threadIdx.x & 31;
  const int wid  = threadIdx.x >> 5;
  const int m0 = blockIdx.x * 256 + wid * 64;
  const int n0 = blockIdx.y * 64;
  if (m0 >= M) return;

  f32x8 acc[4][4];
#pragma unroll
  for (int i = 0; i < 4; ++i)
#pragma unroll
    for (int j = 0; j < 4; ++j)
#pragma unroll
      for (int r = 0; r < 8; ++r) acc[i][j][r] = 0.0f;

  for (int k0 = 0; k0 < K; k0 += 32) {
    bf16x16 a[4];
#pragma unroll
    for (int i = 0; i < 4; ++i) a[i] = load_a_frag(A, K, m0 + i * 16, k0, lane);
#pragma unroll
    for (int j = 0; j < 4; ++j) {
      bf16x16 b = load_b_frag(Bt, K, n0 + j * 16, k0, lane);
#pragma unroll
      for (int i = 0; i < 4; ++i) acc[i][j] = wmma_bf16(a[i], b, acc[i][j]);
    }
  }

  const int rowoff  = (lane >> 4) << 3;
  const int colbase = n0 + (lane & 15);
#pragma unroll
  for (int i = 0; i < 4; ++i)
#pragma unroll
    for (int j = 0; j < 4; ++j) {
      int col  = colbase + j * 16;
      float bb = bias ? bias[col] : 0.0f;
#pragma unroll
      for (int r = 0; r < 8; ++r) {
        int row = m0 + i * 16 + rowoff + r;
        float v = acc[i][j][r] + bb;
        if (relu) v = fmaxf(v, 0.0f);
        Y[row * N + col] = (bf16)v;
      }
    }
}

// ---------------------------------------------------------------------------
// Fused LSTM cell: z = XH[M,Kcat] @ Wcat[512,Kcat]^T + bsum, gates i,f,g,o,
//   c' = sig(f)*c + sig(i)*tanh(g);  h' = sig(o)*tanh(c')
// Wave tile = 32 rows x 16 hidden cols, 2x4 gate accumulators per wave
// (8 wmma per K-step vs 12 b128 loads).
// Block = 256 threads = 8 waves (row strips); grid = (M/256, D/16).
// ---------------------------------------------------------------------------
__global__ __launch_bounds__(256)
void lstm_fused_kernel(const bf16* __restrict__ XH, const bf16* __restrict__ Wcat,
                       const float* __restrict__ bsum, float* __restrict__ Cst,
                       bf16* __restrict__ Hst, int M, int Kcat) {
  const int lane = threadIdx.x & 31;
  const int wid  = threadIdx.x >> 5;
  const int m0 = blockIdx.x * 256 + wid * 32;
  const int n0 = blockIdx.y * 16;  // within hidden dim D
  if (m0 >= M) return;

  f32x8 acc[2][4];  // [m-subtile][gate i,f,g,o]
#pragma unroll
  for (int i = 0; i < 2; ++i)
#pragma unroll
    for (int g = 0; g < 4; ++g)
#pragma unroll
      for (int r = 0; r < 8; ++r) acc[i][g][r] = 0.0f;

  for (int k0 = 0; k0 < Kcat; k0 += 32) {
    bf16x16 a0 = load_a_frag(XH, Kcat, m0,      k0, lane);
    bf16x16 a1 = load_a_frag(XH, Kcat, m0 + 16, k0, lane);
#pragma unroll
    for (int g = 0; g < 4; ++g) {
      bf16x16 b = load_b_frag(Wcat, Kcat, g * D_DIM + n0, k0, lane);
      acc[0][g] = wmma_bf16(a0, b, acc[0][g]);
      acc[1][g] = wmma_bf16(a1, b, acc[1][g]);
    }
  }

  const int col = n0 + (lane & 15);
  const float bi_ = bsum[0 * D_DIM + col];
  const float bf_ = bsum[1 * D_DIM + col];
  const float bg_ = bsum[2 * D_DIM + col];
  const float bo_ = bsum[3 * D_DIM + col];
  const int rowoff = (lane >> 4) << 3;
#pragma unroll
  for (int i = 0; i < 2; ++i)
#pragma unroll
    for (int r = 0; r < 8; ++r) {
      int row  = m0 + i * 16 + rowoff + r;
      int idx  = row * D_DIM + col;
      float zi = acc[i][0][r] + bi_;
      float zf = acc[i][1][r] + bf_;
      float zg = acc[i][2][r] + bg_;
      float zo = acc[i][3][r] + bo_;
      float cn = sigmoidf_(zf) * Cst[idx] + sigmoidf_(zi) * tanhf(zg);
      float hn = sigmoidf_(zo) * tanhf(cn);
      Cst[idx] = cn;
      Hst[idx] = (bf16)hn;
    }
}

// ---------------------------------------------------------------------------
// Small support kernels
// ---------------------------------------------------------------------------
__global__ void zero_f32_kernel(float* p, int n) {
  int i = blockIdx.x * blockDim.x + threadIdx.x;
  if (i < n) p[i] = 0.0f;
}

__global__ void init_state_kernel(const float* __restrict__ hinit,
                                  bf16* __restrict__ H, float* __restrict__ C,
                                  int rows) {
  int idx = blockIdx.x * blockDim.x + threadIdx.x;
  if (idx >= rows * D_DIM) return;
  H[idx] = (bf16)hinit[idx & (D_DIM - 1)];
  C[idx] = 0.0f;
}

// Wt[n*K+k] = bf16(W[k*N+n])  -- transpose to [N][K] for B-fragment loads
__global__ void convT_kernel(const float* __restrict__ W, bf16* __restrict__ Wt,
                             int K, int N) {
  int n = blockIdx.x, k = threadIdx.x;
  Wt[n * K + k] = (bf16)W[k * N + n];
}

// Wcat[n*Kcat+k] = k<in_f ? Wih[n][k] : Whh[n][k-in_f]  (rows already [N][K])
__global__ void conv_lstm_kernel(const float* __restrict__ Wih,
                                 const float* __restrict__ Whh,
                                 bf16* __restrict__ Wcat, int in_f, int Kcat) {
  int n = blockIdx.x, k = threadIdx.x;
  float v = (k < in_f) ? Wih[n * in_f + k] : Whh[n * D_DIM + (k - in_f)];
  Wcat[n * Kcat + k] = (bf16)v;
}

__global__ void bias_sum_kernel(const float* a, const float* b, float* s) {
  int n = threadIdx.x;
  s[n] = a[n] + b[n];
}

// segment-sum: one block per edge, 128 lanes add one feature each
__global__ void scatter_add_kernel(const bf16* __restrict__ P,
                                   const int* __restrict__ srcIdx,
                                   const int* __restrict__ dstIdx,
                                   float* __restrict__ out) {
  int e = blockIdx.x, d = threadIdx.x;
  float v = (float)P[srcIdx[e] * D_DIM + d];
  atomicAdd(out + dstIdx[e] * D_DIM + d, v);
}

// xh = [bf16(seg) | Ch]  (Kcat = 256)
__global__ void pack_c_kernel(const float* __restrict__ seg,
                              const bf16* __restrict__ Hc,
                              bf16* __restrict__ xh) {
  int m = blockIdx.x, d = threadIdx.x;
  xh[m * 256 + d]         = (bf16)seg[m * D_DIM + d];
  xh[m * 256 + D_DIM + d] = Hc[m * D_DIM + d];
}

// xh = [bf16(seg) | flip(Lh) | Lh]  (Kcat = 384)
__global__ void pack_l_kernel(const float* __restrict__ seg,
                              const bf16* __restrict__ Hl,
                              bf16* __restrict__ xh) {
  int m = blockIdx.x, d = threadIdx.x;
  xh[m * 384 + d]             = (bf16)seg[m * D_DIM + d];
  xh[m * 384 + D_DIM + d]     = Hl[(NLITS - 1 - m) * D_DIM + d];
  xh[m * 384 + 2 * D_DIM + d] = Hl[m * D_DIM + d];
}

// votes[i] = dot(h2[i], w) + b  (final 128->1 layer)
__global__ void vote_dot_kernel(const bf16* __restrict__ X,
                                const float* __restrict__ w,
                                const float* __restrict__ b,
                                float* __restrict__ votes, int M) {
  int i = blockIdx.x * blockDim.x + threadIdx.x;
  if (i >= M) return;
  const bf16* row = X + i * D_DIM;
  float s = 0.0f;
#pragma unroll
  for (int d = 0; d < D_DIM; ++d) s += (float)row[d] * w[d];
  votes[i] = s + b[0];
}

// logits[b][j] = sum_v votes[j*NVARS + b*500 + v] + vote_bias
__global__ void reduce_votes_kernel(const float* __restrict__ votes,
                                    const float* __restrict__ vote_bias,
                                    float* __restrict__ out) {
  int t = threadIdx.x;
  if (t >= 2 * N_BATCH) return;
  int b = t >> 1, j = t & 1;
  const int per = NVARS / N_BATCH;  // 500
  float s = 0.0f;
  for (int v = 0; v < per; ++v) s += votes[j * NVARS + b * per + v];
  out[b * 2 + j] = s + vote_bias[0];
}

// ---------------------------------------------------------------------------
// Host orchestration
// ---------------------------------------------------------------------------
extern "C" void kernel_launch(void* const* d_in, const int* in_sizes, int n_in,
                              void* d_out, int out_size, void* d_ws, size_t ws_size,
                              hipStream_t stream) {
  // ---- inputs (setup_inputs dict order, tuples flattened) ----
  const int*   edge_clause = (const int*)d_in[0];
  const int*   edge_lit    = (const int*)d_in[1];
  // d_in[2] is_sat (unused), d_in[3..5] scalars n_vars/n_lits/n_clauses (hardcoded)
  const float* L_init = (const float*)d_in[6];
  const float* C_init = (const float*)d_in[7];
  const float* lcW[3] = {(const float*)d_in[8],  (const float*)d_in[10], (const float*)d_in[12]};
  const float* lcB[3] = {(const float*)d_in[9],  (const float*)d_in[11], (const float*)d_in[13]};
  const float* clW[3] = {(const float*)d_in[14], (const float*)d_in[16], (const float*)d_in[18]};
  const float* clB[3] = {(const float*)d_in[15], (const float*)d_in[17], (const float*)d_in[19]};
  const float* lWih = (const float*)d_in[20];
  const float* lWhh = (const float*)d_in[21];
  const float* lbih = (const float*)d_in[22];
  const float* lbhh = (const float*)d_in[23];
  const float* cWih = (const float*)d_in[24];
  const float* cWhh = (const float*)d_in[25];
  const float* cbih = (const float*)d_in[26];
  const float* cbhh = (const float*)d_in[27];
  const float* vW[3] = {(const float*)d_in[28], (const float*)d_in[30], (const float*)d_in[32]};
  const float* vB[3] = {(const float*)d_in[29], (const float*)d_in[31], (const float*)d_in[33]};
  const float* vote_bias = (const float*)d_in[34];
  float* out = (float*)d_out;

  // ---- workspace carve-out ----
  char*  ws  = (char*)d_ws;
  size_t off = 0;
  auto carve = [&](size_t bytes) -> void* {
    void* p = ws + off;
    off = (off + bytes + 255) & ~(size_t)255;
    return p;
  };
  bf16*  Lh   = (bf16*)carve((size_t)NLITS    * D_DIM * sizeof(bf16));
  float* Lc   = (float*)carve((size_t)NLITS    * D_DIM * sizeof(float));
  bf16*  Ch   = (bf16*)carve((size_t)NCLAUSES * D_DIM * sizeof(bf16));
  float* Cc   = (float*)carve((size_t)NCLAUSES * D_DIM * sizeof(float));
  bf16*  act0 = (bf16*)carve((size_t)NCLAUSES * D_DIM * sizeof(bf16));
  bf16*  act1 = (bf16*)carve((size_t)NCLAUSES * D_DIM * sizeof(bf16));
  float* seg  = (float*)carve((size_t)NCLAUSES * D_DIM * sizeof(float));
  bf16*  xh   = (bf16*)carve((size_t)NCLAUSES * 256 * sizeof(bf16));  // >= 32000*384
  float* votes = (float*)carve((size_t)NLITS * sizeof(float));
  bf16*  lcWt  = (bf16*)carve((size_t)3 * D_DIM * D_DIM * sizeof(bf16));
  bf16*  clWt  = (bf16*)carve((size_t)3 * D_DIM * D_DIM * sizeof(bf16));
  bf16*  vWt   = (bf16*)carve((size_t)2 * D_DIM * D_DIM * sizeof(bf16));
  bf16*  cWcat = (bf16*)carve((size_t)512 * 256 * sizeof(bf16));
  bf16*  lWcat = (bf16*)carve((size_t)512 * 384 * sizeof(bf16));
  float* cbsum = (float*)carve(512 * sizeof(float));
  float* lbsum = (float*)carve(512 * sizeof(float));
  (void)ws_size; (void)in_sizes; (void)n_in; (void)out_size;

  // ---- one-time (per launch) weight conversion ----
  for (int l = 0; l < 3; ++l) {
    convT_kernel<<<dim3(D_DIM), dim3(D_DIM), 0, stream>>>(lcW[l], lcWt + l * D_DIM * D_DIM, D_DIM, D_DIM);
    convT_kernel<<<dim3(D_DIM), dim3(D_DIM), 0, stream>>>(clW[l], clWt + l * D_DIM * D_DIM, D_DIM, D_DIM);
  }
  convT_kernel<<<dim3(D_DIM), dim3(D_DIM), 0, stream>>>(vW[0], vWt,                 D_DIM, D_DIM);
  convT_kernel<<<dim3(D_DIM), dim3(D_DIM), 0, stream>>>(vW[1], vWt + D_DIM * D_DIM, D_DIM, D_DIM);
  conv_lstm_kernel<<<dim3(512), dim3(256), 0, stream>>>(cWih, cWhh, cWcat, 128, 256);
  conv_lstm_kernel<<<dim3(512), dim3(384), 0, stream>>>(lWih, lWhh, lWcat, 256, 384);
  bias_sum_kernel<<<dim3(1), dim3(512), 0, stream>>>(cbih, cbhh, cbsum);
  bias_sum_kernel<<<dim3(1), dim3(512), 0, stream>>>(lbih, lbhh, lbsum);

  // ---- init states ----
  init_state_kernel<<<dim3((NLITS * D_DIM + 255) / 256), dim3(256), 0, stream>>>(L_init, Lh, Lc, NLITS);
  init_state_kernel<<<dim3((NCLAUSES * D_DIM + 255) / 256), dim3(256), 0, stream>>>(C_init, Ch, Cc, NCLAUSES);

  auto gemm = [&](const bf16* A, const bf16* Bt, const float* bias, bf16* Y,
                  int M, int K, int N, int relu) {
    gemm_bf16_kernel<<<dim3(M / 256, N / 64), dim3(128), 0, stream>>>(A, Bt, bias, Y, M, K, N, relu);
  };

  // ---- message-passing rounds ----
  for (int r = 0; r < NROUNDS; ++r) {
    // literal -> clause: lc_pre = MLP(Lh)
    gemm(Lh,   lcWt,                     lcB[0], act0, NLITS, D_DIM, D_DIM, 1);
    gemm(act0, lcWt + 1 * D_DIM * D_DIM, lcB[1], act1, NLITS, D_DIM, D_DIM, 1);
    gemm(act1, lcWt + 2 * D_DIM * D_DIM, lcB[2], act0, NLITS, D_DIM, D_DIM, 0);
    {
      int n = NCLAUSES * D_DIM;
      zero_f32_kernel<<<dim3((n + 255) / 256), dim3(256), 0, stream>>>(seg, n);
    }
    scatter_add_kernel<<<dim3(NEDGES), dim3(D_DIM), 0, stream>>>(act0, edge_lit, edge_clause, seg);
    pack_c_kernel<<<dim3(NCLAUSES), dim3(D_DIM), 0, stream>>>(seg, Ch, xh);
    lstm_fused_kernel<<<dim3(NCLAUSES / 256, D_DIM / 16), dim3(256), 0, stream>>>(
        xh, cWcat, cbsum, Cc, Ch, NCLAUSES, 256);

    // clause -> literal: cl_pre = MLP(Ch)
    gemm(Ch,   clWt,                     clB[0], act0, NCLAUSES, D_DIM, D_DIM, 1);
    gemm(act0, clWt + 1 * D_DIM * D_DIM, clB[1], act1, NCLAUSES, D_DIM, D_DIM, 1);
    gemm(act1, clWt + 2 * D_DIM * D_DIM, clB[2], act0, NCLAUSES, D_DIM, D_DIM, 0);
    {
      int n = NLITS * D_DIM;
      zero_f32_kernel<<<dim3((n + 255) / 256), dim3(256), 0, stream>>>(seg, n);
    }
    scatter_add_kernel<<<dim3(NEDGES), dim3(D_DIM), 0, stream>>>(act0, edge_clause, edge_lit, seg);
    pack_l_kernel<<<dim3(NLITS), dim3(D_DIM), 0, stream>>>(seg, Lh, xh);
    lstm_fused_kernel<<<dim3(NLITS / 256, D_DIM / 16), dim3(256), 0, stream>>>(
        xh, lWcat, lbsum, Lc, Lh, NLITS, 384);
  }

  // ---- vote head ----
  gemm(Lh,   vWt,                 vB[0], act0, NLITS, D_DIM, D_DIM, 1);
  gemm(act0, vWt + D_DIM * D_DIM, vB[1], act1, NLITS, D_DIM, D_DIM, 1);
  vote_dot_kernel<<<dim3((NLITS + 255) / 256), dim3(256), 0, stream>>>(act1, vW[2], vB[2], votes, NLITS);
  reduce_votes_kernel<<<dim3(1), dim3(64), 0, stream>>>(votes, vote_bias, out);
}
